// RowSelfAttention_42082089566794
// MI455X (gfx1250) — matrix operand order repaired and checked by reading, not compiled, not run on hardware
//
#include <hip/hip_runtime.h>
#include <hip/hip_bf16.h>
#include <math.h>

// Problem constants (match reference)
#define RR 512
#define CC 512
#define EE 256
#define HH 8
#define DD 32

typedef __attribute__((ext_vector_type(16))) __bf16          v16bf;
typedef __attribute__((ext_vector_type(8)))  float           v8f;
typedef __attribute__((ext_vector_type(16))) unsigned short  v16us;
typedef __attribute__((ext_vector_type(4)))  unsigned int    v4u;
typedef __attribute__((ext_vector_type(8)))  unsigned int    v8u;
typedef __attribute__((ext_vector_type(4)))  float           v4f;
typedef __attribute__((ext_vector_type(4)))  unsigned short  v4us;

union FragU {
    v16us us;
    v16bf bf;
    v4u   u4[2];
};

__device__ __forceinline__ unsigned short f32_to_bf16(float f) {
    unsigned int u = __float_as_uint(f);
    u = (u + 0x7FFFu + ((u >> 16) & 1u)) >> 16;   // round to nearest even
    return (unsigned short)u;
}

// A-matrix 16x32 bf16 fragment: lane holds row M=lane&15,
// K = kbase..kbase+15 contiguous, kbase = 16*(lane>>4).
__device__ __forceinline__ v16bf load_frag_a(const unsigned short* base) {
    FragU f;
    f.u4[0] = *(const v4u*)(base);
    f.u4[1] = *(const v4u*)(base + 8);
    return f.bf;
}

// B-matrix 32x16 bf16 fragment: lane holds column N=lane&15,
// K = kstart..kstart+15 contiguous, kstart = 16*(lane>>4).
__device__ __forceinline__ v16bf load_frag_b16(const unsigned short* base) {
    FragU f;
    f.u4[0] = *(const v4u*)(base);
    f.u4[1] = *(const v4u*)(base + 8);
    return f.bf;
}

#define WMMA_BF16(A, B, C) \
    __builtin_amdgcn_wmma_f32_16x16x32_bf16(false, (A), false, (B), (short)0, (C), false, false)

// ---------------------------------------------------------------------------
// Kernel 0: fp32 -> bf16 bulk convert (vectorized)
// ---------------------------------------------------------------------------
__global__ __launch_bounds__(256)
void k_cvt_f32_bf16(const float* __restrict__ s, unsigned short* __restrict__ d, long n) {
    long stride = (long)gridDim.x * blockDim.x * 4;
    for (long t = ((long)blockIdx.x * blockDim.x + threadIdx.x) * 4; t < n; t += stride) {
        v4f v = *(const v4f*)(s + t);
        v4us o;
#pragma unroll
        for (int e = 0; e < 4; ++e) o[e] = f32_to_bf16(v[e]);
        *(v4us*)(d + t) = o;
    }
}

// ---------------------------------------------------------------------------
// Kernel 0b: W [K x N] fp32 -> Wt [N x K] bf16
// ---------------------------------------------------------------------------
__global__ __launch_bounds__(256)
void k_transpose_cvt(const float* __restrict__ W, unsigned short* __restrict__ Wt,
                     int K, int N) {
    int n = blockIdx.x;
    for (int k = threadIdx.x; k < K; k += blockDim.x)
        Wt[(long)n * K + k] = f32_to_bf16(W[(long)k * N + n]);
}

// ---------------------------------------------------------------------------
// Kernel 1/5: Y[M,N] = X[M,K](bf16) @ Wt[N,K]^T + bias; optional q-epilogue.
// One 16(M)x64(N) tile per wave. All 5 fragments (1xA, 4xB) are loaded into
// distinct registers before the WMMA group so load waits can stage instead of
// fully draining before each matrix op.
// ---------------------------------------------------------------------------
__global__ __launch_bounds__(256)
void k_gemm_xw(const unsigned short* __restrict__ X,
               const unsigned short* __restrict__ Wt,
               const float* __restrict__ bias,
               const float* __restrict__ maskRow,   // [M] or nullptr
               float scale,
               unsigned short* __restrict__ Ybf,    // bf16 out (or null)
               float* __restrict__ Yf,              // f32 out (or null)
               int M, int K, int N) {
    const int lane = threadIdx.x & 31;
    const int wave = threadIdx.x >> 5;
    const int tiles64 = N >> 6;                      // N/64
    const long tile = (long)blockIdx.x * 8 + wave;
    const long nTiles = (long)(M >> 4) * tiles64;
    if (tile >= nTiles) return;                      // wave-uniform
    const int tm = (int)(tile / tiles64);
    const int tn = (int)(tile % tiles64);

    const int rowA  = tm * 16 + (lane & 15);
    const int koffA = (lane >> 4) * 16;
    const int koffB = (lane >> 4) * 16;
    const unsigned short* aptr = X + (long)rowA * K + koffA;
    const unsigned short* bp0 = Wt + (long)(tn * 64 +      (lane & 15)) * K + koffB;
    const unsigned short* bp1 = Wt + (long)(tn * 64 + 16 + (lane & 15)) * K + koffB;
    const unsigned short* bp2 = Wt + (long)(tn * 64 + 32 + (lane & 15)) * K + koffB;
    const unsigned short* bp3 = Wt + (long)(tn * 64 + 48 + (lane & 15)) * K + koffB;

    v8f acc0 = {}, acc1 = {}, acc2 = {}, acc3 = {};
    for (int kk = 0; kk < K; kk += 32) {
        __builtin_prefetch((const void*)(aptr + kk + 64), 0, 3);
        v16bf a  = load_frag_a(aptr + kk);
        v16bf b0 = load_frag_b16(bp0 + kk);
        v16bf b1 = load_frag_b16(bp1 + kk);
        v16bf b2 = load_frag_b16(bp2 + kk);
        v16bf b3 = load_frag_b16(bp3 + kk);
        acc0 = WMMA_BF16(a, b0, acc0);
        acc1 = WMMA_BF16(a, b1, acc1);
        acc2 = WMMA_BF16(a, b2, acc2);
        acc3 = WMMA_BF16(a, b3, acc3);
    }

    const int mBase = tm * 16 + 8 * (lane >> 4);
    v8f accs[4] = {acc0, acc1, acc2, acc3};
#pragma unroll
    for (int n = 0; n < 4; ++n) {
        const int   coln = tn * 64 + n * 16 + (lane & 15);
        const float bn   = bias[coln];
#pragma unroll
        for (int v = 0; v < 8; ++v) {
            int   m   = mBase + v;
            float val = accs[n][v] + bn;
            if (maskRow) val *= scale * (1.0f - maskRow[m]);
            if (Yf)  Yf [(long)m * N + coln] = val;
            else     Ybf[(long)m * N + coln] = f32_to_bf16(val);
        }
    }
}

// ---------------------------------------------------------------------------
// Kernel 2: logits[h,i,j] = sum_{r,d} q[r,i,h,d] * k[r,j,h,d]
// 32x32 output tile per wave (2x2 register blocking), K = R*D = 16384.
// ---------------------------------------------------------------------------
__global__ __launch_bounds__(256)
void k_qk_logits(const unsigned short* __restrict__ Q,
                 const unsigned short* __restrict__ Kb,
                 float* __restrict__ logits) {          // [H,C,C] fp32
    const int lane = threadIdx.x & 31;
    const int wave = threadIdx.x >> 5;
    const int tile = blockIdx.x * 8 + wave;             // H*16*16 = 2048 tiles
    const int TJ = CC / 32;
    const int h   = tile / (TJ * TJ);
    const int rem = tile % (TJ * TJ);
    const int ti = rem / TJ, tj = rem % TJ;

    const int i0 = ti * 32 + (lane & 15);
    const int j0 = tj * 32 + (lane & 15);
    const int dA = (lane >> 4) * 16;
    const int dB = (lane >> 4) * 16;

    v8f a00 = {}, a01 = {}, a10 = {}, a11 = {};
    for (int r = 0; r < RR; ++r) {
        const unsigned short* ab0 = Q  + ((long)(r * CC + i0) * EE + h * DD);
        const unsigned short* bb0 = Kb + ((long)(r * CC + j0) * EE + h * DD);
        __builtin_prefetch((const void*)(ab0 + (long)4 * CC * EE), 0, 3);
        __builtin_prefetch((const void*)(bb0 + (long)4 * CC * EE), 0, 3);
        v16bf a0 = load_frag_a(ab0 + dA);
        v16bf a1 = load_frag_a(ab0 + 16 * EE + dA);
        v16bf b0 = load_frag_b16(bb0 + dB);
        v16bf b1 = load_frag_b16(bb0 + 16 * EE + dB);
        a00 = WMMA_BF16(a0, b0, a00);
        a01 = WMMA_BF16(a0, b1, a01);
        a10 = WMMA_BF16(a1, b0, a10);
        a11 = WMMA_BF16(a1, b1, a11);
    }

    const int iB0 = ti * 32 + 8 * (lane >> 4);
    const int iB1 = iB0 + 16;
    const int jc0 = tj * 32 + (lane & 15);
    const int jc1 = jc0 + 16;
#pragma unroll
    for (int v = 0; v < 8; ++v) {
        float* r0 = logits + ((long)h * CC + (iB0 + v)) * CC;
        float* r1 = logits + ((long)h * CC + (iB1 + v)) * CC;
        r0[jc0] = a00[v];  r0[jc1] = a01[v];
        r1[jc0] = a10[v];  r1[jc1] = a11[v];
    }
}

// ---------------------------------------------------------------------------
// Kernel 3: key-padding mask + softmax over j; one wave per (h,i) row.
// ---------------------------------------------------------------------------
__global__ __launch_bounds__(256)
void k_mask_softmax(const float* __restrict__ logits,
                    const float* __restrict__ mask0,     // mask[0][:]
                    float* __restrict__ probsF,
                    unsigned short* __restrict__ probsB) {
    const int  lane = threadIdx.x & 31;
    const int  wave = threadIdx.x >> 5;
    const long row  = (long)blockIdx.x * 8 + wave;       // H*C rows
    const float* src = logits + row * CC;

    float vals[CC / 32];
    float mx = -3.0e38f;
#pragma unroll
    for (int t = 0; t < CC / 32; ++t) {
        int   jj = t * 32 + lane;
        float m0 = mask0[jj];
        float w  = src[jj];
        w = w * (1.0f - m0) + m0 * (-10000.0f);
        vals[t] = w;
        mx = fmaxf(mx, w);
    }
#pragma unroll
    for (int o = 16; o >= 1; o >>= 1) mx = fmaxf(mx, __shfl_xor(mx, o, 32));

    float sum = 0.0f;
#pragma unroll
    for (int t = 0; t < CC / 32; ++t) {
        vals[t] = __expf(vals[t] - mx);
        sum += vals[t];
    }
#pragma unroll
    for (int o = 16; o >= 1; o >>= 1) sum += __shfl_xor(sum, o, 32);
    const float inv = 1.0f / sum;

#pragma unroll
    for (int t = 0; t < CC / 32; ++t) {
        int   jj = t * 32 + lane;
        float p  = vals[t] * inv;
        probsF[row * CC + jj] = p;
        probsB[row * CC + jj] = f32_to_bf16(p);
    }
}

// ---------------------------------------------------------------------------
// Kernel 4: context[r,i,h,d] = sum_j probs[h,i,j] * v[r,j,h,d]
// 32(i) x 32(d) tile per wave per (h,r). The j-major V tile is DMA'd into LDS
// by the Tensor Data Mover with DOUBLE BUFFERING: the TDM for step k+1 is
// issued before consuming step k, and we wait with s_wait_tensorcnt <= 1 so
// the DMA overlaps the 4 WMMAs + gathers. pad_enable (1 dword per 16) makes
// the column gathers bank-conflict-free.
// ---------------------------------------------------------------------------
__global__ __launch_bounds__(256)
void k_attn_context(const unsigned short* __restrict__ P,   // [H,C,C] bf16
                    const unsigned short* __restrict__ V,   // [R*C,E] bf16
                    unsigned short* __restrict__ Ctx) {     // [R*C,E] bf16
    // per wave: 2 buffers x 32 rows x (16 data + 1 pad) dwords = 1088 dwords
    __shared__ unsigned int ldsV[8][2 * 32 * 17];

    const int lane = threadIdx.x & 31;
    const int wave = threadIdx.x >> 5;
    const int tile = blockIdx.x * 8 + wave;        // H*R*(C/32) = 65536
    const int TI = CC / 32;
    const int ti = tile % TI;
    int t1 = tile / TI;
    const int r = t1 % RR;
    const int h = t1 / RR;

    const int kA = (lane >> 4) * 16;
    const int kB = (lane >> 4) * 16;
    const unsigned short* arow0 = P + ((long)h * CC + ti * 32 + (lane & 15)) * CC + kA;
    const unsigned short* arow1 = arow0 + 16 * CC;

    const unsigned int ldsBase = (unsigned int)(wave * (2 * 32 * 17 * 4));
    const unsigned int bufOff[2] = { ldsBase, ldsBase + 32 * 17 * 4 };

    // D# group 1 (constant): data_size=2B, pad_enable, pad_interval=16 dwords,
    // pad_amount=1 dword, tensor_dim0=E, tensor_dim1=R*C, tile=32x32,
    // tensor_dim0_stride=E.
    v8u g1;
    g1[0] = (1u << 16) | (1u << 20) | (3u << 22);
    g1[1] = ((unsigned)EE & 0xFFFFu) << 16;
    g1[2] = (((unsigned)EE >> 16) & 0xFFFFu) | (((unsigned)(RR * CC) & 0xFFFFu) << 16);
    g1[3] = (((unsigned)(RR * CC) >> 16) & 0xFFFFu) | (32u << 16);
    g1[4] = 32u;
    g1[5] = (unsigned)EE;
    g1[6] = 0u;
    g1[7] = 0u;

    // helper: issue one TDM 32x32 tile load (rows kk..kk+31 of head h)
    auto issue_tdm = [&](unsigned int ldsOff, int kk) {
        const unsigned short* gptr = V + ((long)(r * CC + kk) * EE + h * DD);
        unsigned long long ga = (unsigned long long)gptr;
        v4u g0;
        g0[0] = 1u;                                   // count=1 (D# group 0)
        g0[1] = ldsOff;
        g0[2] = (unsigned int)ga;
        g0[3] = (unsigned int)((ga >> 32) & 0x01FFFFFFull) | 0x80000000u; // type=2
        asm volatile("tensor_load_to_lds %0, %1" :: "s"(g0), "s"(g1) : "memory");
    };

    v8f acc00 = {}, acc01 = {}, acc10 = {}, acc11 = {};

    issue_tdm(bufOff[0], 0);                          // prologue
    for (int kk = 0; kk < CC; kk += 32) {
        const int cur = (kk >> 5) & 1;
        if (kk + 32 < CC) {
            // gathers from buf[cur^1] (previous iter) must drain before overwrite
            asm volatile("s_wait_dscnt 0x0" ::: "memory");
            issue_tdm(bufOff[cur ^ 1], kk + 32);
            __builtin_amdgcn_s_wait_tensorcnt(1);     // buf[cur] ready, next in flight
        } else {
            __builtin_amdgcn_s_wait_tensorcnt(0);
        }

        const unsigned short* lbase =
            (const unsigned short*)&ldsV[0][0] + (bufOff[cur] >> 1);
        FragU b0, b1;
#pragma unroll
        for (int t = 0; t < 16; ++t) {
            const int rowo = (kB + t) * 34;           // 17 dwords = 34 ushorts pitch
            b0.us[t] = lbase[rowo + (lane & 15)];
            b1.us[t] = lbase[rowo + 16 + (lane & 15)];
        }
        v16bf a0 = load_frag_a(arow0 + kk);
        v16bf a1 = load_frag_a(arow1 + kk);
        acc00 = WMMA_BF16(a0, b0.bf, acc00);
        acc01 = WMMA_BF16(a0, b1.bf, acc01);
        acc10 = WMMA_BF16(a1, b0.bf, acc10);
        acc11 = WMMA_BF16(a1, b1.bf, acc11);
    }

    const int iB0 = ti * 32 + 8 * (lane >> 4);
    const int iB1 = iB0 + 16;
    const int d0  = (lane & 15);
    const int d1  = 16 + (lane & 15);
#pragma unroll
    for (int v = 0; v < 8; ++v) {
        long base0 = ((long)(r * CC + iB0 + v)) * EE + h * DD;
        long base1 = ((long)(r * CC + iB1 + v)) * EE + h * DD;
        Ctx[base0 + d0] = f32_to_bf16(acc00[v]);
        Ctx[base0 + d1] = f32_to_bf16(acc01[v]);
        Ctx[base1 + d0] = f32_to_bf16(acc10[v]);
        Ctx[base1 + d1] = f32_to_bf16(acc11[v]);
    }
}

// ---------------------------------------------------------------------------
extern "C" void kernel_launch(void* const* d_in, const int* in_sizes, int n_in,
                              void* d_out, int out_size, void* d_ws, size_t ws_size,
                              hipStream_t stream) {
    (void)in_sizes; (void)n_in; (void)out_size; (void)ws_size;

    const float* x    = (const float*)d_in[0];
    const float* mask = (const float*)d_in[1];
    const float* Wq   = (const float*)d_in[2];
    const float* bq   = (const float*)d_in[3];
    const float* Wk   = (const float*)d_in[4];
    const float* bk   = (const float*)d_in[5];
    const float* Wv   = (const float*)d_in[6];
    const float* bv   = (const float*)d_in[7];
    const float* Wo   = (const float*)d_in[8];
    const float* bo   = (const float*)d_in[9];

    float* out      = (float*)d_out;                       // [R*C, E] fp32
    float* attn_out = out + (long)RR * CC * EE;            // [H, C, C] fp32

    // workspace layout
    char*  ws = (char*)d_ws;
    const size_t SZ = (size_t)RR * CC * EE * 2;            // one bf16 tensor
    unsigned short* xb  = (unsigned short*)(ws);
    unsigned short* qb  = (unsigned short*)(ws + SZ);
    unsigned short* kb  = (unsigned short*)(ws + 2 * SZ);
    unsigned short* vb  = (unsigned short*)(ws + 3 * SZ);
    unsigned short* wt  = (unsigned short*)(ws + 4 * SZ);  // 4 * E*E bf16
    unsigned short* WqT = wt;
    unsigned short* WkT = wt + (size_t)EE * EE;
    unsigned short* WvT = wt + 2 * (size_t)EE * EE;
    unsigned short* WoT = wt + 3 * (size_t)EE * EE;
    float*          logits = (float*)(ws + 4 * SZ + 4 * (size_t)EE * EE * 2);
    unsigned short* pbf    = (unsigned short*)((char*)logits + (size_t)HH * CC * CC * 4);
    unsigned short* ctx    = qb;                           // reuse q after logits

    const float scaling = 1.0f / (sqrtf((float)DD) * sqrtf((float)RR));
    const int   M = RR * CC, K = EE, N = EE;

    // 0) convert x to bf16; pre-transpose + convert weights
    k_cvt_f32_bf16<<<8192, 256, 0, stream>>>(x, xb, (long)RR * CC * EE);
    k_transpose_cvt<<<EE, 256, 0, stream>>>(Wq, WqT, EE, EE);
    k_transpose_cvt<<<EE, 256, 0, stream>>>(Wk, WkT, EE, EE);
    k_transpose_cvt<<<EE, 256, 0, stream>>>(Wv, WvT, EE, EE);
    k_transpose_cvt<<<EE, 256, 0, stream>>>(Wo, WoT, EE, EE);

    // 1) q/k/v projections (q gets scaling * (1-mask) epilogue)
    const int gemmBlocks = (M / 16) * (N / 64) / 8;        // 8192
    k_gemm_xw<<<gemmBlocks, 256, 0, stream>>>(xb, WqT, bq, mask, scaling,
                                              qb, nullptr, M, K, N);
    k_gemm_xw<<<gemmBlocks, 256, 0, stream>>>(xb, WkT, bk, nullptr, 1.0f,
                                              kb, nullptr, M, K, N);
    k_gemm_xw<<<gemmBlocks, 256, 0, stream>>>(xb, WvT, bv, nullptr, 1.0f,
                                              vb, nullptr, M, K, N);

    // 2) row-summed attention logits  [H,C,C]
    k_qk_logits<<<HH * (CC / 32) * (CC / 32) / 8, 256, 0, stream>>>(qb, kb, logits);

    // 3) key-padding mask + softmax -> fp32 probs (output #2) + bf16 copy
    k_mask_softmax<<<HH * CC / 8, 256, 0, stream>>>(logits, mask, attn_out, pbf);

    // 4) context = probs @ v  (double-buffered TDM-staged V tiles)
    k_attn_context<<<HH * RR * (CC / 32) / 8, 256, 0, stream>>>(pbf, vb, ctx);

    // 5) output projection (fp32 store into d_out)
    k_gemm_xw<<<gemmBlocks, 256, 0, stream>>>(ctx, WoT, bo, nullptr, 1.0f,
                                              nullptr, out, M, K, N);
}
